// BottleneckA_73143293051463
// MI455X (gfx1250) — matrix operand order repaired and checked by reading, not compiled, not run on hardware
//
#include <hip/hip_runtime.h>

// ---------------------------------------------------------------------------
// Fused MoE bottleneck for MI455X (gfx1250, wave32, WMMA).
//   h1 = relu((x  @ W1^T + b1) * relu(gate))   [per batch: W1(256x1024) * x(1024x3136)]
//   h2 = relu((h1 @ W2^T + b2) * relu(gate))
//   out = relu(h2 @ W3^T + b3) + x
// bf16 operands / f32 accumulate via v_wmma_f32_16x16x32_bf16.
//
// Operand VGPR layouts (CDNA5 ISA 7.12.2, wave32):
//   A 16x32 bf16 : lane = row m (lane%16); K interleaved in 8-groups by lane
//                  half: half0 -> K{0..7,16..23}, half1 -> K{8..15,24..31}.
//   B 32x16 bf16 : lane = col n (lane%16); K interleaved in 8-groups by lane
//                  half: half0 -> K{0..7,16..23}, half1 -> K{8..15,24..31}.
//   C/D 16x16 f32: lane = col n (lane%16); VGPR v -> row m = 8*(lane/16)+v.
// ---------------------------------------------------------------------------

#define C_DIM    1024
#define MID      256
#define NSP      3136          // 56*56
#define NB       16            // batch
#define TB       32            // spatial columns per workgroup
#define XSTR     (C_DIM + 8)   // padded LDS row stride (bf16 elems) for x tile
#define HSTR     (MID + 8)     // padded LDS row stride for h1/h2 tiles
#define NTHREADS 512
#define NWAVES   16

#define W1_ELEMS (MID * C_DIM)     // 262144
#define W2_ELEMS (MID * MID)       // 65536
#define W3_ELEMS (C_DIM * MID)     // 262144
#define W_TOTAL  (W1_ELEMS + W2_ELEMS + W3_ELEMS)  // 589824

typedef __attribute__((ext_vector_type(16))) __bf16 v16bf;
typedef __attribute__((ext_vector_type(8)))  float  v8f;

union ABU { v16bf v; uint4 q[2]; };

__device__ __forceinline__ unsigned short f2bf(float f) {
  unsigned u = __float_as_uint(f);
  u += 0x7FFFu + ((u >> 16) & 1u);       // round-to-nearest-even
  return (unsigned short)(u >> 16);
}

__device__ __forceinline__ uint4 pack8(const float* f) {
  uint4 q;
  q.x = (unsigned)f2bf(f[0]) | ((unsigned)f2bf(f[1]) << 16);
  q.y = (unsigned)f2bf(f[2]) | ((unsigned)f2bf(f[3]) << 16);
  q.z = (unsigned)f2bf(f[4]) | ((unsigned)f2bf(f[5]) << 16);
  q.w = (unsigned)f2bf(f[6]) | ((unsigned)f2bf(f[7]) << 16);
  return q;
}

// ---------------- weight fp32 -> bf16 conversion (runs once per call) ------
__global__ void cvt_weights(const float* __restrict__ W1,
                            const float* __restrict__ W2,
                            const float* __restrict__ W3,
                            unsigned short* __restrict__ out) {
  int i = blockIdx.x * blockDim.x + threadIdx.x;
  if (i < W1_ELEMS)                      out[i] = f2bf(W1[i]);
  else if (i < W1_ELEMS + W2_ELEMS)      out[i] = f2bf(W2[i - W1_ELEMS]);
  else if (i < W_TOTAL)                  out[i] = f2bf(W3[i - (W1_ELEMS + W2_ELEMS)]);
}

// ---------------- fused main kernel ----------------------------------------
__global__ __launch_bounds__(NTHREADS) void moe_fused(
    const float* __restrict__ x, const float* __restrict__ gate,
    const unsigned short* __restrict__ Wbf,
    const float* __restrict__ b1, const float* __restrict__ b2,
    const float* __restrict__ b3, float* __restrict__ out)
{
  // LDS: x tile [n][k] bf16 (transposed, padded), h1/h2 tiles [n][m] bf16.
  __shared__ unsigned short xT [TB * XSTR];   // 66048 B
  __shared__ unsigned short h1T[TB * HSTR];   // 16896 B
  __shared__ unsigned short h2T[TB * HSTR];   // 16896 B  -> 99840 B total

  const int b    = blockIdx.y;
  const int n0   = blockIdx.x * TB;
  const int tid  = threadIdx.x;
  const int wave = tid >> 5;
  const int lane = tid & 31;
  const int half = lane >> 4;
  const int r    = lane & 15;   // A: row m within tile / B,C,D: column n within tile

  const unsigned short* W1bf = Wbf;
  const unsigned short* W2bf = Wbf + W1_ELEMS;
  const unsigned short* W3bf = Wbf + W1_ELEMS + W2_ELEMS;

  // ---- stage x tile: global [k][n] fp32 (n contiguous) -> LDS [n][k] bf16 ----
  const float* xb = x + (size_t)b * C_DIM * NSP + n0;
  #pragma unroll
  for (int it = 0; it < (C_DIM * TB) / (NTHREADS * 4); ++it) {   // 16 iters
    int e  = it * NTHREADS + tid;
    int k  = e >> 3;              // 0..1023
    int n4 = (e & 7) << 2;        // 0,4,...,28
    float4 v = *(const float4*)(xb + (size_t)k * NSP + n4);
    xT[(n4 + 0) * XSTR + k] = f2bf(v.x);
    xT[(n4 + 1) * XSTR + k] = f2bf(v.y);
    xT[(n4 + 2) * XSTR + k] = f2bf(v.z);
    xT[(n4 + 3) * XSTR + k] = f2bf(v.w);
  }
  __syncthreads();

  // GEMM of one 16-wide M strip against both 16-wide N tiles of this block.
  // Arow: lane's weight row (bf16, k contiguous). Bt: LDS [n][k] operand tile.
  auto gemm2n = [&](const unsigned short* Arow, int K,
                    const unsigned short* Bt, int bstr, v8f& a0, v8f& a1) {
    for (int k0 = 0; k0 < K; k0 += 32) {
      ABU a, bb0, bb1;
      // A: elements 0..7 -> K = k0 + half*8 + {0..7}; 8..15 -> +16
      const unsigned short* ap = Arow + k0 + half * 8;
      a.q[0] = *(const uint4*)(ap);
      a.q[1] = *(const uint4*)(ap + 16);
      // B: elements 0..7 -> K = k0 + half*8 + {0..7}; 8..15 -> +16
      const unsigned short* bp0 = Bt + (size_t)r * bstr + k0 + half * 8;
      bb0.q[0] = *(const uint4*)(bp0);
      bb0.q[1] = *(const uint4*)(bp0 + 16);
      const unsigned short* bp1 = bp0 + 16 * bstr;
      bb1.q[0] = *(const uint4*)(bp1);
      bb1.q[1] = *(const uint4*)(bp1 + 16);
      a0 = __builtin_amdgcn_wmma_f32_16x16x32_bf16(false, a.v, false, bb0.v,
                                                   (short)0, a0, false, false);
      a1 = __builtin_amdgcn_wmma_f32_16x16x32_bf16(false, a.v, false, bb1.v,
                                                   (short)0, a1, false, false);
    }
  };

  const float* gb = gate + (size_t)b * MID;

  // ---- GEMM1: h1[256 x TB] = relu((W1 * x + b1) * relu(gate)) -> LDS bf16 --
  {
    const int m0 = wave * 16;
    v8f a0 = {}, a1 = {};
    gemm2n(W1bf + (size_t)(m0 + r) * C_DIM, C_DIM, xT, XSTR, a0, a1);
    const int mb = m0 + half * 8;               // 8 consecutive output channels
    float4 bA = *(const float4*)(b1 + mb);
    float4 bB = *(const float4*)(b1 + mb + 4);
    float4 gA = *(const float4*)(gb + mb);
    float4 gB = *(const float4*)(gb + mb + 4);
    float bias[8] = {bA.x, bA.y, bA.z, bA.w, bB.x, bB.y, bB.z, bB.w};
    float g[8]    = {fmaxf(gA.x,0.f), fmaxf(gA.y,0.f), fmaxf(gA.z,0.f), fmaxf(gA.w,0.f),
                     fmaxf(gB.x,0.f), fmaxf(gB.y,0.f), fmaxf(gB.z,0.f), fmaxf(gB.w,0.f)};
    float f[8];
    #pragma unroll
    for (int v = 0; v < 8; ++v) f[v] = fmaxf((a0[v] + bias[v]) * g[v], 0.f);
    *(uint4*)&h1T[r * HSTR + mb] = pack8(f);
    #pragma unroll
    for (int v = 0; v < 8; ++v) f[v] = fmaxf((a1[v] + bias[v]) * g[v], 0.f);
    *(uint4*)&h1T[(16 + r) * HSTR + mb] = pack8(f);
  }
  __syncthreads();

  // ---- GEMM2: h2 = relu((W2 * h1 + b2) * relu(gate)) -> LDS bf16 ----------
  {
    const int m0 = wave * 16;
    v8f a0 = {}, a1 = {};
    gemm2n(W2bf + (size_t)(m0 + r) * MID, MID, h1T, HSTR, a0, a1);
    const int mb = m0 + half * 8;
    float4 bA = *(const float4*)(b2 + mb);
    float4 bB = *(const float4*)(b2 + mb + 4);
    float4 gA = *(const float4*)(gb + mb);
    float4 gB = *(const float4*)(gb + mb + 4);
    float bias[8] = {bA.x, bA.y, bA.z, bA.w, bB.x, bB.y, bB.z, bB.w};
    float g[8]    = {fmaxf(gA.x,0.f), fmaxf(gA.y,0.f), fmaxf(gA.z,0.f), fmaxf(gA.w,0.f),
                     fmaxf(gB.x,0.f), fmaxf(gB.y,0.f), fmaxf(gB.z,0.f), fmaxf(gB.w,0.f)};
    float f[8];
    #pragma unroll
    for (int v = 0; v < 8; ++v) f[v] = fmaxf((a0[v] + bias[v]) * g[v], 0.f);
    *(uint4*)&h2T[r * HSTR + mb] = pack8(f);
    #pragma unroll
    for (int v = 0; v < 8; ++v) f[v] = fmaxf((a1[v] + bias[v]) * g[v], 0.f);
    *(uint4*)&h2T[(16 + r) * HSTR + mb] = pack8(f);
  }
  __syncthreads();

  // ---- GEMM3: out = relu(W3 * h2 + b3) + x (4 M strips per wave) ----------
  for (int s = 0; s < 4; ++s) {
    const int m0 = (s * NWAVES + wave) * 16;
    v8f a0 = {}, a1 = {};
    gemm2n(W3bf + (size_t)(m0 + r) * MID, MID, h2T, HSTR, a0, a1);
    const int mb = m0 + half * 8;
    float4 bA = *(const float4*)(b3 + mb);
    float4 bB = *(const float4*)(b3 + mb + 4);
    float bias[8] = {bA.x, bA.y, bA.z, bA.w, bB.x, bB.y, bB.z, bB.w};
    #pragma unroll
    for (int v = 0; v < 8; ++v) {
      size_t gi = ((size_t)b * C_DIM + (mb + v)) * NSP + (n0 + r);
      out[gi] = fmaxf(a0[v] + bias[v], 0.f) + x[gi];
    }
    #pragma unroll
    for (int v = 0; v < 8; ++v) {
      size_t gi = ((size_t)b * C_DIM + (mb + v)) * NSP + (n0 + 16 + r);
      out[gi] = fmaxf(a1[v] + bias[v], 0.f) + x[gi];
    }
  }
}

// ---------------------------------------------------------------------------
extern "C" void kernel_launch(void* const* d_in, const int* in_sizes, int n_in,
                              void* d_out, int out_size, void* d_ws, size_t ws_size,
                              hipStream_t stream) {
  const float* x    = (const float*)d_in[0];
  const float* gate = (const float*)d_in[1];
  const float* W1   = (const float*)d_in[2];
  const float* b1   = (const float*)d_in[3];
  const float* W2   = (const float*)d_in[4];
  const float* b2   = (const float*)d_in[5];
  const float* W3   = (const float*)d_in[6];
  const float* b3   = (const float*)d_in[7];
  float* out = (float*)d_out;
  unsigned short* Wbf = (unsigned short*)d_ws;   // needs W_TOTAL*2 = 1.18 MB

  cvt_weights<<<W_TOTAL / 256, 256, 0, stream>>>(W1, W2, W3, Wbf);

  dim3 grid(NSP / TB, NB);                       // 98 x 16 workgroups
  moe_fused<<<grid, NTHREADS, 0, stream>>>(x, gate, Wbf, b1, b2, b3, out);
}
